// SelfAttention_26980984554036
// MI455X (gfx1250) — compile-verified, matrix operand order checked by reading, not run on hardware
//
#include <hip/hip_runtime.h>
#include <stdint.h>

// ---------------------------------------------------------------------------
// Prefix-LM self-attention for MI455X (gfx1250, wave32, WMMA bf16 + TDM).
//   B=4, S=2048, D=1024, H=16, dh=64
// Pipeline: qkv_gemm (fp32 x @ fp32 W -> bf16 q/k/v [B,H,S,dh])
//           attn_kernel (flash attention, bf16 WMMA, TDM-staged K tiles)
//           proj_gemm (bf16 attn @ fp32 Wp + bp -> fp32 out, TDM-staged A)
// ---------------------------------------------------------------------------

#define SEQ 2048
#define DM  1024
#define NH  16
#define DH  64
#define NB  4
#define HEAD_ELEMS ((size_t)SEQ * DH)          // per (b,h) elements
#define TENSOR_ELEMS ((size_t)NB * SEQ * DM)   // 8388608 elements per q/k/v

typedef __bf16 bf16x16 __attribute__((ext_vector_type(16)));
typedef float floatx8 __attribute__((ext_vector_type(8)));
typedef uint32_t u32x4 __attribute__((ext_vector_type(4)));
typedef uint32_t u32x8 __attribute__((ext_vector_type(8)));

union ABFrag { bf16x16 v; uint32_t u[8]; };

// round-to-nearest-even fp32 -> bf16 (bit trick, pure VALU)
__device__ __forceinline__ uint16_t bf16_1(float x) {
  uint32_t a = __float_as_uint(x);
  return (uint16_t)((a + 0x7FFFu + ((a >> 16) & 1u)) >> 16);
}
__device__ __forceinline__ uint32_t pk_bf16(float lo, float hi) {
  return (uint32_t)bf16_1(lo) | ((uint32_t)bf16_1(hi) << 16);
}

// Load one 16x32 bf16 operand fragment (A layout; B tiles are stored
// transposed in LDS so the same loader works with N playing the role of M).
// ISA 7.12.2: lane m = lane&15; lanes 16-31 shift K by +8;
// VGPR v holds K pair: v0..3 -> K{0,2,4,6}, v4..7 -> K{16,18,20,22} (+half*8).
__device__ __forceinline__ bf16x16 load_frag(const uint32_t* base, int rstride_dw) {
  const int lane = threadIdx.x & 31;
  const uint32_t* p = base + (lane & 15) * rstride_dw + ((lane >> 4) << 2);
  ABFrag f;
  f.u[0] = p[0];  f.u[1] = p[1];  f.u[2] = p[2];  f.u[3] = p[3];
  f.u[4] = p[8];  f.u[5] = p[9];  f.u[6] = p[10]; f.u[7] = p[11];
  return f.v;
}

__device__ __forceinline__ floatx8 wmma_bf16(bf16x16 a, bf16x16 b, floatx8 c) {
  return __builtin_amdgcn_wmma_f32_16x16x32_bf16(false, a, false, b,
                                                 (short)0, c, false, false);
}

// ---------------------------------------------------------------------------
// Tensor Data Mover: 2D tile load global -> LDS (ISA ch.8 D# layout).
// data_size = 8B units; optional LDS row padding via pad_interval/pad_amount.
// group0: {count=1, lds_addr, global_addr[56:0], type=2}
// group1: {flags, tensor_dim0/1, tile_dim0/1, tensor_dim0_stride}
// All inputs must be wave-uniform; issued once per wave (EXEC ignored).
// ---------------------------------------------------------------------------
__device__ __forceinline__ void tdm_load_2d(uint64_t gaddr, uint32_t lds_off,
                                            uint32_t pad_interval_code,
                                            uint32_t pad_amount_code,
                                            uint32_t tensor_d0, uint32_t tensor_d1,
                                            uint32_t tile_d0, uint32_t tile_d1,
                                            uint32_t stride0_units) {
  u32x4 g0;
  g0[0] = 1u;                                     // count=1, user mode
  g0[1] = lds_off;                                // LDS byte address
  g0[2] = (uint32_t)gaddr;                        // global_addr[31:0]
  g0[3] = (uint32_t)(gaddr >> 32) | 0x80000000u;  // global_addr[56:32] | type=2
  u32x8 g1;
  g1[0] = (3u << 16)                              // data_size = 8 bytes
        | (1u << 20)                              // pad_enable
        | (pad_interval_code << 22)
        | (pad_amount_code << 25);
  g1[1] = (tensor_d0 & 0xFFFFu) << 16;            // tensor_dim0[15:0]
  g1[2] = (tensor_d0 >> 16) | ((tensor_d1 & 0xFFFFu) << 16);
  g1[3] = (tensor_d1 >> 16) | (tile_d0 << 16);    // tile_dim0
  g1[4] = tile_d1;                                // tile_dim1 (tile_dim2 = 0)
  g1[5] = stride0_units;                          // tensor_dim0_stride[31:0]
  g1[6] = 0u;
  g1[7] = 0u;
  asm volatile("tensor_load_to_lds %0, %1" :: "s"(g0), "s"(g1) : "memory");
}

// ---------------------------------------------------------------------------
// Kernel 1: QKV projections. grid = (64 Mtiles, 16 Ntiles, 3), block = 256.
// Block tile 128(M) x 64(N), K-step 32. 8 waves in 4x2, each 32x32 output.
// (fp32 -> bf16 conversion during staging, so no TDM here.)
// ---------------------------------------------------------------------------
__global__ __launch_bounds__(256) void qkv_gemm(
    const float* __restrict__ x,
    const float* __restrict__ Wq, const float* __restrict__ bq,
    const float* __restrict__ Wk, const float* __restrict__ bk,
    const float* __restrict__ Wv, const float* __restrict__ bv,
    uint16_t* __restrict__ ws) {
  const int which = blockIdx.z;
  const float* W    = (which == 0) ? Wq : (which == 1) ? Wk : Wv;
  const float* bias = (which == 0) ? bq : (which == 1) ? bk : bv;
  uint16_t* dst = ws + (size_t)which * TENSOR_ELEMS;

  __shared__ uint32_t sA[128 * 18];  // 128 rows x 32 bf16 (16 dw) + 2 dw pad
  __shared__ uint32_t sB[64 * 18];   // 64 n-rows x 32 bf16 (transposed W tile)

  const int lane = threadIdx.x & 31;
  const int wave = threadIdx.x >> 5;
  const int wm = wave >> 1;          // 0..3
  const int wn = wave & 1;           // 0..1
  const int row0 = blockIdx.x * 128;
  const int n0   = blockIdx.y * 64;

  floatx8 acc[2][2] = {};

  for (int kt = 0; kt < DM; kt += 32) {
    // ---- stage A: x[row0..+128][kt..+32] fp32 -> bf16 LDS (4 float4/thr)
#pragma unroll
    for (int i = 0; i < 4; ++i) {
      int idx = threadIdx.x + 256 * i;          // 0..1023
      int r = idx >> 3, c4 = idx & 7;
      const float4 f = *(const float4*)(x + (size_t)(row0 + r) * DM + kt + c4 * 4);
      sA[r * 18 + c4 * 2]     = pk_bf16(f.x, f.y);
      sA[r * 18 + c4 * 2 + 1] = pk_bf16(f.z, f.w);
    }
    // ---- stage B transposed: W[kt..+32][n0..+64] -> sB[n][k] (2 float4/thr)
    uint16_t* b16 = (uint16_t*)sB;
#pragma unroll
    for (int i = 0; i < 2; ++i) {
      int idx = threadIdx.x + 256 * i;          // 0..511
      int kr = idx >> 4, c4 = idx & 15;
      const float4 f = *(const float4*)(W + (size_t)(kt + kr) * DM + n0 + c4 * 4);
      b16[(c4 * 4 + 0) * 36 + kr] = bf16_1(f.x);
      b16[(c4 * 4 + 1) * 36 + kr] = bf16_1(f.y);
      b16[(c4 * 4 + 2) * 36 + kr] = bf16_1(f.z);
      b16[(c4 * 4 + 3) * 36 + kr] = bf16_1(f.w);
    }
    __syncthreads();

    bf16x16 a0 = load_frag(sA + (wm * 32)      * 18, 18);
    bf16x16 a1 = load_frag(sA + (wm * 32 + 16) * 18, 18);
    bf16x16 b0 = load_frag(sB + (wn * 32)      * 18, 18);
    bf16x16 b1 = load_frag(sB + (wn * 32 + 16) * 18, 18);
    acc[0][0] = wmma_bf16(a0, b0, acc[0][0]);
    acc[0][1] = wmma_bf16(a0, b1, acc[0][1]);
    acc[1][0] = wmma_bf16(a1, b0, acc[1][0]);
    acc[1][1] = wmma_bf16(a1, b1, acc[1][1]);
    __syncthreads();
  }

  // ---- epilogue: + bias, write bf16 in [B, H, S, dh] layout
  const int half = lane >> 4, n = lane & 15;
#pragma unroll
  for (int a = 0; a < 2; ++a)
#pragma unroll
    for (int bb = 0; bb < 2; ++bb) {
      const int col = n0 + wn * 32 + bb * 16 + n;
      const float bval = bias[col];
      const int h = col >> 6, d = col & 63;
#pragma unroll
      for (int r = 0; r < 8; ++r) {
        const int row = row0 + wm * 32 + a * 16 + r + 8 * half;
        const int bi = row >> 11, si = row & (SEQ - 1);
        const float v = acc[a][bb][r] + bval;
        dst[(((size_t)bi * NH + h) * SEQ + si) * DH + d] = bf16_1(v);
      }
    }
}

// ---------------------------------------------------------------------------
// Kernel 2: flash attention. grid = (16 qtiles, 16 heads, 4 batch), block=256.
// Each wave owns 16 query rows; 32-key tiles: K tile staged by TDM (wave 0,
// padded rows: 16x8B + 8B pad -> 34-dword stride), V transposed by all waves.
// ---------------------------------------------------------------------------
__global__ __launch_bounds__(256) void attn_kernel(
    const uint16_t* __restrict__ qws, const uint16_t* __restrict__ kws,
    const uint16_t* __restrict__ vws, const int* __restrict__ plen_arr,
    uint16_t* __restrict__ attn_out) {
  const int b = blockIdx.z, h = blockIdx.y;
  const int q0 = blockIdx.x * 128;
  const int lane = threadIdx.x & 31;
  const int wave = threadIdx.x >> 5;
  const int qrow0 = q0 + wave * 16;

  __shared__ uint32_t sK[32 * 34];       // 32 keys x 64 d (32 dw + 2 pad)
  __shared__ uint32_t sV[64 * 18];       // 64 d-rows x 32 keys (transposed V)
  __shared__ uint32_t sP[8 * 16 * 18];   // per-wave P tile 16x32

  const size_t headBase = ((size_t)(b * NH + h)) * HEAD_ELEMS;   // elements
  const uint32_t* qg = (const uint32_t*)qws + ((headBase + (size_t)qrow0 * DH) >> 1);
  // Q fragments live in registers for the whole key loop (d 0..31 / 32..63)
  const bf16x16 qa0 = load_frag(qg, 32);
  const bf16x16 qa1 = load_frag(qg + 16, 32);

  const int plen = plen_arr[b];
  int kend = q0 + 128;
  if (plen > kend) kend = plen;
  kend = (kend + 31) & ~31;
  if (kend > SEQ) kend = SEQ;

  float rm[8], rl[8];
#pragma unroll
  for (int r = 0; r < 8; ++r) { rm[r] = -1e30f; rl[r] = 0.0f; }
  floatx8 acc[4] = {};

  const uint64_t kbase_byte = (uint64_t)(uintptr_t)kws + 2ull * headBase;
  const uint32_t ldsK = (uint32_t)(uintptr_t)(void*)sK;
  const uint32_t* vsrc = (const uint32_t*)vws + (headBase >> 1);
  const bool tdmWave = (threadIdx.x < 32);
  const int half = lane >> 4, nn = lane & 15;

  for (int kt = 0; kt < kend; kt += 32) {
    // ---- TDM: K tile [32 keys][64 bf16] -> sK with 8B pad per 128B row.
    // units = 8B: tile 16 x 32, stride 16, pad_interval code 4 (16 units),
    // pad_amount code 1 (2 dwords) => LDS row stride 34 dwords.
    if (tdmWave)
      tdm_load_2d(kbase_byte + (uint64_t)kt * (DH * 2), ldsK,
                  /*pad_interval*/4u, /*pad_amount*/1u,
                  /*tensor_d0*/16u, /*tensor_d1*/32u,
                  /*tile_d0*/16u, /*tile_d1*/32u, /*stride0*/16u);

    // ---- stage V tile transposed -> sV[d][key] (all waves, overlaps TDM)
    uint16_t* v16 = (uint16_t*)sV;
#pragma unroll
    for (int i = 0; i < 4; ++i) {
      int idx = threadIdx.x + 256 * i;
      int key = idx >> 5, c = idx & 31;    // dword c holds d=2c,2c+1
      uint32_t val = vsrc[(size_t)(kt + key) * 32 + c];
      v16[(2 * c) * 36 + key]     = (uint16_t)val;
      v16[(2 * c + 1) * 36 + key] = (uint16_t)(val >> 16);
    }
    // prefetch next key tile (global_prefetch_b8)
    if (kt + 32 < kend)
      __builtin_prefetch((const uint32_t*)vsrc + (size_t)(kt + 32) * 32 +
                             (threadIdx.x & 31), 0, 0);
    if (tdmWave) __builtin_amdgcn_s_wait_tensorcnt(0);
    __syncthreads();

    // ---- S = Q K^T (16 q-rows x 32 keys), K-dim = 64 in 2 steps
    floatx8 sc0 = {}, sc1 = {};
    {
      bf16x16 kb0 = load_frag(sK, 34);              // keys 0-15,  d 0..31
      bf16x16 kb1 = load_frag(sK + 16 * 34, 34);    // keys 16-31, d 0..31
      sc0 = wmma_bf16(qa0, kb0, sc0);
      sc1 = wmma_bf16(qa0, kb1, sc1);
      kb0 = load_frag(sK + 16, 34);                 // d 32..63
      kb1 = load_frag(sK + 16 * 34 + 16, 34);
      sc0 = wmma_bf16(qa1, kb0, sc0);
      sc1 = wmma_bf16(qa1, kb1, sc1);
    }

    // ---- mask + online softmax (C layout: n = lane&15, m = r + 8*half)
    const int j0 = kt + nn, j1 = kt + 16 + nn;
    float pr0[8], pr1[8];
#pragma unroll
    for (int r = 0; r < 8; ++r) {
      const int i = qrow0 + r + 8 * half;
      float s0 = sc0[r] * 0.125f;
      float s1 = sc1[r] * 0.125f;
      if (!((j0 <= i) || (j0 < plen))) s0 = -1e30f;
      if (!((j1 <= i) || (j1 < plen))) s1 = -1e30f;
      float mx = fmaxf(s0, s1);
#pragma unroll
      for (int m = 1; m <= 8; m <<= 1) mx = fmaxf(mx, __shfl_xor(mx, m, 32));
      const float nm = fmaxf(rm[r], mx);
      const float corr = __expf(rm[r] - nm);
      const float p0 = __expf(s0 - nm);
      const float p1 = __expf(s1 - nm);
      float sum = p0 + p1;
#pragma unroll
      for (int m = 1; m <= 8; m <<= 1) sum += __shfl_xor(sum, m, 32);
      rl[r] = rl[r] * corr + sum;
      rm[r] = nm;
#pragma unroll
      for (int f = 0; f < 4; ++f) acc[f][r] *= corr;
      pr0[r] = p0;
      pr1[r] = p1;
    }

    // ---- C-layout P -> A-layout via per-wave LDS round trip
    uint16_t* p16 = (uint16_t*)(sP + wave * 16 * 18);
#pragma unroll
    for (int r = 0; r < 8; ++r) {
      const int m = r + 8 * half;
      p16[m * 36 + nn]      = bf16_1(pr0[r]);
      p16[m * 36 + 16 + nn] = bf16_1(pr1[r]);
    }
    asm volatile("s_wait_dscnt 0" ::: "memory");  // same-wave LDS RAW
    const bf16x16 pa = load_frag(sP + wave * 16 * 18, 18);

    // ---- O += P V  (4 d-fragments of 16)
#pragma unroll
    for (int f = 0; f < 4; ++f) {
      bf16x16 vb = load_frag(sV + (f * 16) * 18, 18);
      acc[f] = wmma_bf16(pa, vb, acc[f]);
    }
    __syncthreads();
  }

  // ---- normalize + store bf16 in [B, S, H*dh] layout for the projection
#pragma unroll
  for (int f = 0; f < 4; ++f)
#pragma unroll
    for (int r = 0; r < 8; ++r) {
      const int row = qrow0 + r + 8 * half;
      const int col = h * DH + f * 16 + nn;
      const float v = acc[f][r] * (1.0f / rl[r]);
      attn_out[((size_t)(b * SEQ + row)) * DM + col] = bf16_1(v);
    }
}

// ---------------------------------------------------------------------------
// Kernel 3: output projection. grid = (64, 16, 1), block = 256.
// A tile (already bf16) staged by TDM; B tile converted fp32->bf16 manually.
// ---------------------------------------------------------------------------
__global__ __launch_bounds__(256) void proj_gemm(
    const uint16_t* __restrict__ attn, const float* __restrict__ Wp,
    const float* __restrict__ bp, float* __restrict__ out) {
  __shared__ uint32_t sA[128 * 18];
  __shared__ uint32_t sB[64 * 18];

  const int lane = threadIdx.x & 31;
  const int wave = threadIdx.x >> 5;
  const int wm = wave >> 1;
  const int wn = wave & 1;
  const int row0 = blockIdx.x * 128;
  const int n0   = blockIdx.y * 64;

  const uint64_t abase_byte =
      (uint64_t)(uintptr_t)attn + 2ull * (size_t)row0 * DM;
  const uint32_t ldsA = (uint32_t)(uintptr_t)(void*)sA;
  const bool tdmWave = (threadIdx.x < 32);

  floatx8 acc[2][2] = {};

  for (int kt = 0; kt < DM; kt += 32) {
    // ---- TDM: A tile [128 rows][32 bf16] -> sA, 8B pad per 64B row.
    // units = 8B: tile 8 x 128, row stride 256 units (1024 bf16),
    // pad_interval code 3 (8 units), pad_amount code 1 => 18-dword stride.
    if (tdmWave)
      tdm_load_2d(abase_byte + (uint64_t)kt * 2, ldsA,
                  /*pad_interval*/3u, /*pad_amount*/1u,
                  /*tensor_d0*/8u, /*tensor_d1*/128u,
                  /*tile_d0*/8u, /*tile_d1*/128u, /*stride0*/256u);

    // ---- stage B transposed from Wp (fp32 -> bf16), overlaps TDM
    uint16_t* b16 = (uint16_t*)sB;
#pragma unroll
    for (int i = 0; i < 2; ++i) {
      int idx = threadIdx.x + 256 * i;
      int kr = idx >> 4, c4 = idx & 15;
      const float4 f = *(const float4*)(Wp + (size_t)(kt + kr) * DM + n0 + c4 * 4);
      b16[(c4 * 4 + 0) * 36 + kr] = bf16_1(f.x);
      b16[(c4 * 4 + 1) * 36 + kr] = bf16_1(f.y);
      b16[(c4 * 4 + 2) * 36 + kr] = bf16_1(f.z);
      b16[(c4 * 4 + 3) * 36 + kr] = bf16_1(f.w);
    }
    if (tdmWave) __builtin_amdgcn_s_wait_tensorcnt(0);
    __syncthreads();

    bf16x16 a0 = load_frag(sA + (wm * 32)      * 18, 18);
    bf16x16 a1 = load_frag(sA + (wm * 32 + 16) * 18, 18);
    bf16x16 b0 = load_frag(sB + (wn * 32)      * 18, 18);
    bf16x16 b1 = load_frag(sB + (wn * 32 + 16) * 18, 18);
    acc[0][0] = wmma_bf16(a0, b0, acc[0][0]);
    acc[0][1] = wmma_bf16(a0, b1, acc[0][1]);
    acc[1][0] = wmma_bf16(a1, b0, acc[1][0]);
    acc[1][1] = wmma_bf16(a1, b1, acc[1][1]);
    __syncthreads();
  }

  const int half = lane >> 4, n = lane & 15;
#pragma unroll
  for (int a = 0; a < 2; ++a)
#pragma unroll
    for (int bb = 0; bb < 2; ++bb) {
      const int col = n0 + wn * 32 + bb * 16 + n;
      const float bval = bp[col];
#pragma unroll
      for (int r = 0; r < 8; ++r) {
        const int row = row0 + wm * 32 + a * 16 + r + 8 * half;
        out[(size_t)row * DM + col] = acc[a][bb][r] + bval;
      }
    }
}

// ---------------------------------------------------------------------------
extern "C" void kernel_launch(void* const* d_in, const int* in_sizes, int n_in,
                              void* d_out, int out_size, void* d_ws, size_t ws_size,
                              hipStream_t stream) {
  (void)in_sizes; (void)n_in; (void)out_size; (void)ws_size;
  const float* x    = (const float*)d_in[0];
  const int*   plen = (const int*)d_in[1];
  const float* Wq = (const float*)d_in[2];  const float* bq = (const float*)d_in[3];
  const float* Wk = (const float*)d_in[4];  const float* bk = (const float*)d_in[5];
  const float* Wv = (const float*)d_in[6];  const float* bv = (const float*)d_in[7];
  const float* Wp = (const float*)d_in[8];  const float* bp = (const float*)d_in[9];

  uint16_t* ws = (uint16_t*)d_ws;
  uint16_t* qw = ws;
  uint16_t* kw = ws + TENSOR_ELEMS;
  uint16_t* vw = ws + 2 * TENSOR_ELEMS;
  uint16_t* aw = ws + 3 * TENSOR_ELEMS;
  float* out = (float*)d_out;

  dim3 g1(64, 16, 3);
  qkv_gemm<<<g1, 256, 0, stream>>>(x, Wq, bq, Wk, bk, Wv, bv, ws);

  dim3 g2(16, 16, 4);
  attn_kernel<<<g2, 256, 0, stream>>>(qw, kw, vw, plen, aw);

  dim3 g3(64, 16, 1);
  proj_gemm<<<g3, 256, 0, stream>>>(aw, Wp, bp, out);
}
